// SpatialTransformer_51049981281055
// MI455X (gfx1250) — matrix operand order repaired and checked by reading, not compiled, not run on hardware
//
#include <hip/hip_runtime.h>
#include <hip/hip_bf16.h>
#include <math.h>

// ---------------------------------------------------------------------------
// SpatialTransformer block for MI455X (gfx1250, wave32, WMMA 16x16x32 f16)
// GEMMs use async global->LDS DMA double-buffering when the toolchain exposes
// the gfx1250 builtin (guarded with __has_builtin; falls back to sync loads).
// ---------------------------------------------------------------------------

typedef _Float16 h16;
typedef __attribute__((ext_vector_type(8)))  _Float16 v8h;
typedef __attribute__((ext_vector_type(16))) _Float16 v16h;
typedef __attribute__((ext_vector_type(8)))  float    v8f;
typedef __attribute__((ext_vector_type(4)))  int      v4i;

#define NTOK  4096     // H*W
#define CDIM  512
#define FFDIM 2048
#define HEADS 8
#define DH    64
#define NCTX  77
#define CTXD  768

#if defined(__gfx1250__) && __has_builtin(__builtin_amdgcn_global_load_async_to_lds_b128) && __has_builtin(__builtin_amdgcn_s_wait_asynccnt)
#define USE_ASYNC_LDS 1
#else
#define USE_ASYNC_LDS 0
#endif

static __device__ __forceinline__ v16h cat16(v8h lo, v8h hi) {
  return __builtin_shufflevector(lo, hi, 0,1,2,3,4,5,6,7,8,9,10,11,12,13,14,15);
}
static __device__ __forceinline__ v8f wmma_f16(v16h a, v16h b, v8f c) {
  // D = A(16x32 f16) x B(32x16 f16) + C(16x16 f32)
  return __builtin_amdgcn_wmma_f32_16x16x32_f16(false, a, false, b, (short)0, c,
                                                false, false);
}
static __device__ __forceinline__ v8h zero8h() {
  v8h z;
#pragma unroll
  for (int i = 0; i < 8; ++i) z[i] = (h16)0;
  return z;
}
static __device__ __forceinline__ v8f zero8f() {
  v8f z;
#pragma unroll
  for (int i = 0; i < 8; ++i) z[i] = 0.0f;
  return z;
}

#if USE_ASYNC_LDS
typedef __attribute__((address_space(1))) v4i* gv4i_p;   // global int4*
typedef __attribute__((address_space(3))) v4i* lv4i_p;   // LDS int4*
static __device__ __forceinline__ void async_ld_b128(const void* g, void* l) {
  __builtin_amdgcn_global_load_async_to_lds_b128(
      (gv4i_p)(void*)g, (lv4i_p)l, 0, 0);
}
#endif

// ---------------------------------------------------------------------------
// GroupNorm (32 groups over [C=512, N=4096]); two-pass
// ---------------------------------------------------------------------------
__global__ __launch_bounds__(256) void gn_stats_kernel(
    const float* __restrict__ x, float* __restrict__ stats) {
  __shared__ float ssum[256], ssq[256];
  const int g = blockIdx.x;                 // group id, 0..31
  const int tid = threadIdx.x;
  const float* p = x + (size_t)g * (16 * NTOK);   // 16 channels * 4096, contiguous
  float s = 0.f, q = 0.f;
  for (int i = tid; i < 16 * NTOK; i += 256) {
    float v = p[i];
    s += v; q += v * v;
  }
  ssum[tid] = s; ssq[tid] = q;
  __syncthreads();
  for (int st = 128; st > 0; st >>= 1) {
    if (tid < st) { ssum[tid] += ssum[tid + st]; ssq[tid] += ssq[tid + st]; }
    __syncthreads();
  }
  if (tid == 0) {
    float mean = ssum[0] * (1.0f / (16 * NTOK));
    float var  = ssq[0] * (1.0f / (16 * NTOK)) - mean * mean;
    stats[g]      = mean;
    stats[32 + g] = rsqrtf(var + 1e-5f);
  }
}

__global__ __launch_bounds__(256) void gn_apply_kernel(
    const float* __restrict__ x, const float* __restrict__ gg,
    const float* __restrict__ gb, const float* __restrict__ stats,
    h16* __restrict__ T) {
  int idx = blockIdx.x * 256 + threadIdx.x;     // idx = n*512 + c
  if (idx >= NTOK * CDIM) return;
  int n = idx >> 9, c = idx & 511;
  int grp = c >> 4;
  float v = x[(size_t)c * NTOK + n];            // transpose [C,N] -> [N,C]
  v = (v - stats[grp]) * stats[32 + grp] * gg[c] + gb[c];
  T[idx] = (h16)v;
}

// ---------------------------------------------------------------------------
// LayerNorm: fp32 [N,512] -> f16 [N,512]
// ---------------------------------------------------------------------------
__global__ __launch_bounds__(256) void ln_kernel(
    const float* __restrict__ Y, const float* __restrict__ g,
    const float* __restrict__ b, h16* __restrict__ T) {
  __shared__ float sm[256], sv[256];
  const int n = blockIdx.x, tid = threadIdx.x;
  const float* row = Y + (size_t)n * CDIM;
  float a0 = row[tid], a1 = row[tid + 256];
  sm[tid] = a0 + a1; sv[tid] = a0 * a0 + a1 * a1;
  __syncthreads();
  for (int st = 128; st > 0; st >>= 1) {
    if (tid < st) { sm[tid] += sm[tid + st]; sv[tid] += sv[tid + st]; }
    __syncthreads();
  }
  float mean = sm[0] * (1.0f / CDIM);
  float var  = sv[0] * (1.0f / CDIM) - mean * mean;
  float rstd = rsqrtf(var + 1e-5f);
  T[(size_t)n * CDIM + tid]       = (h16)((a0 - mean) * rstd * g[tid] + b[tid]);
  T[(size_t)n * CDIM + tid + 256] = (h16)((a1 - mean) * rstd * g[tid + 256] + b[tid + 256]);
}

// ---------------------------------------------------------------------------
// Elementwise converters
// ---------------------------------------------------------------------------
__global__ __launch_bounds__(256) void cvt_f32_f16_kernel(
    const float* __restrict__ in, h16* __restrict__ out, int n) {
  int i = blockIdx.x * 256 + threadIdx.x;
  if (i < n) out[i] = (h16)in[i];
}
// weight [K,D] fp32 -> [D,K] f16 (transpose so GEMM B-tiles load contiguous rows)
__global__ __launch_bounds__(256) void wcvt_t_kernel(
    const float* __restrict__ in, h16* __restrict__ out, int K, int D) {
  int idx = blockIdx.x * 256 + threadIdx.x;
  if (idx >= K * D) return;
  int d = idx / K, k = idx - d * K;
  out[idx] = (h16)in[(size_t)k * D + d];
}

// ---------------------------------------------------------------------------
// Tiled WMMA GEMM:  out[M,N] = A[M,K](f16) * Bt[N,K](f16) (+bias)(+act)(+resid)
// Block 256 threads = 8 waves, tile 128x64, BK=32, each wave 32x32 (4 WMMAs).
// Async path: double-buffered LDS tiles filled by GLOBAL_LOAD_ASYNC_TO_LDS_B128,
// stage k+1 DMA overlapped with stage k WMMA, synced via s_wait_asynccnt.
// ACT: 0 none, 1 exact GELU
// OMODE: 0 f16 out [M,N]; 1 f32 out [M,N] (+resid same layout);
//        2 f32 out transposed [N,M] (+resid transposed) -- for proj_out
// ---------------------------------------------------------------------------
template <int ACT, int OMODE>
__global__ __launch_bounds__(256) void gemm_wmma_kernel(
    const h16* __restrict__ A, const h16* __restrict__ Bt,
    const float* __restrict__ bias, const float* __restrict__ resid,
    h16* __restrict__ outH, float* __restrict__ outF,
    int M, int Ndim, int K) {
#if USE_ASYNC_LDS
  __shared__ __align__(16) h16 As[2][128 * 40];  // stride 40 halves (80B, 16B-aligned)
  __shared__ __align__(16) h16 Bs[2][64 * 40];
#else
  __shared__ __align__(16) h16 As[1][128 * 40];
  __shared__ __align__(16) h16 Bs[1][64 * 40];
#endif
  const int tid  = threadIdx.x;
  const int lane = tid & 31;
  const int wave = tid >> 5;
  const int wm = wave & 3;        // 4 waves along M
  const int wn = wave >> 2;       // 2 waves along N
  const int m0 = blockIdx.x * 128;
  const int n0 = blockIdx.y * 64;

  v8f acc[2][2];
#pragma unroll
  for (int mi = 0; mi < 2; ++mi)
#pragma unroll
    for (int ni = 0; ni < 2; ++ni) acc[mi][ni] = zero8f();

  const int arow = tid >> 1, ahk = (tid & 1) * 16;   // 128 rows, 16 halves each
  const int brow = tid >> 2, bhk = (tid & 3) * 8;    // 64 rows, 8 halves each
  const int hs = (lane < 16) ? 0 : 8;                // A-frag K split
  const int kb = (lane < 16) ? 0 : 8;                // B-frag K split
  const int nk = K >> 5;

#if USE_ASYNC_LDS
  // OOB rows: clamp source address (ragged-M outputs are discarded in epilogue)
  const int gra = (m0 + arow < M) ? (m0 + arow) : (M - 1);
  const int grb = (n0 + brow < Ndim) ? (n0 + brow) : (Ndim - 1);
  const h16* aRow = A  + (size_t)gra * K;
  const h16* bRow = Bt + (size_t)grb * K;
  auto issue = [&](int kbIdx, int buf) {
    const int k0 = kbIdx * 32;
    async_ld_b128(aRow + k0 + ahk,     &As[buf][arow * 40 + ahk]);
    async_ld_b128(aRow + k0 + ahk + 8, &As[buf][arow * 40 + ahk + 8]);
    async_ld_b128(bRow + k0 + bhk,     &Bs[buf][brow * 40 + bhk]);
  };
  issue(0, 0);
#endif

  for (int kbi = 0; kbi < nk; ++kbi) {
#if USE_ASYNC_LDS
    const int cur = kbi & 1;
    if (kbi + 1 < nk) {
      issue(kbi + 1, cur ^ 1);                 // overlap next-stage DMA
      __builtin_amdgcn_s_wait_asynccnt(3);     // wait for current stage only
    } else {
      __builtin_amdgcn_s_wait_asynccnt(0);
    }
    __syncthreads();                           // all waves' stage data visible
#else
    const int cur = 0;
    const int k0 = kbi * 32;
    __syncthreads();
#pragma unroll
    for (int i = 0; i < 2; ++i) {
      v8h t = zero8h();
      int gr = m0 + arow;
      if (gr < M) t = *(const v8h*)(A + (size_t)gr * K + k0 + ahk + i * 8);
      *(v8h*)(&As[0][arow * 40 + ahk + i * 8]) = t;
    }
    {
      v8h t = zero8h();
      int gr = n0 + brow;
      if (gr < Ndim) t = *(const v8h*)(Bt + (size_t)gr * K + k0 + bhk);
      *(v8h*)(&Bs[0][brow * 40 + bhk]) = t;
    }
    __syncthreads();
#endif

    v16h af[2], bf[2];
#pragma unroll
    for (int mi = 0; mi < 2; ++mi) {
      int r = wm * 32 + mi * 16 + (lane & 15);
      af[mi] = cat16(*(const v8h*)(&As[cur][r * 40 + hs]),
                     *(const v8h*)(&As[cur][r * 40 + hs + 16]));
    }
#pragma unroll
    for (int ni = 0; ni < 2; ++ni) {
      int r = wn * 32 + ni * 16 + (lane & 15);
      bf[ni] = cat16(*(const v8h*)(&Bs[cur][r * 40 + kb]),
                     *(const v8h*)(&Bs[cur][r * 40 + kb + 16]));
    }
#pragma unroll
    for (int mi = 0; mi < 2; ++mi)
#pragma unroll
      for (int ni = 0; ni < 2; ++ni)
        acc[mi][ni] = wmma_f16(af[mi], bf[ni], acc[mi][ni]);

#if USE_ASYNC_LDS
    __syncthreads();   // everyone done reading buf[cur] before it is re-filled
#endif
  }

  // epilogue (C/D layout: VGPR g, lane<16 -> row g, lane>=16 -> row g+8; col = lane&15)
#pragma unroll
  for (int mi = 0; mi < 2; ++mi) {
#pragma unroll
    for (int ni = 0; ni < 2; ++ni) {
      int col = n0 + wn * 32 + ni * 16 + (lane & 15);
      float bv = bias ? bias[col] : 0.0f;
#pragma unroll
      for (int g = 0; g < 8; ++g) {
        int row = m0 + wm * 32 + mi * 16 + ((lane < 16) ? g : g + 8);
        if (row >= M) continue;
        float v = acc[mi][ni][g] + bv;
        if (ACT == 1) v = 0.5f * v * (1.0f + erff(v * 0.70710678118654752f));
        if (OMODE == 0) {
          outH[(size_t)row * Ndim + col] = (h16)v;
        } else if (OMODE == 1) {
          size_t idx = (size_t)row * Ndim + col;
          if (resid) v += resid[idx];
          outF[idx] = v;
        } else {
          size_t idx = (size_t)col * M + row;   // transposed store [N,M]
          if (resid) v += resid[idx];
          outF[idx] = v;
        }
      }
    }
  }
}

// ---------------------------------------------------------------------------
// Fused flash attention (online softmax), WMMA f16.
// Grid: (N/64, HEADS), block = 128 threads = 4 waves; each wave owns 16 q-rows.
// Q,K,V are f16 [rows, 512] with head h occupying cols h*64..h*64+63.
// Never materializes scores to global memory.
// ---------------------------------------------------------------------------
__global__ __launch_bounds__(128) void attn_flash_kernel(
    const h16* __restrict__ Q, const h16* __restrict__ Kb,
    const h16* __restrict__ Vb, h16* __restrict__ O,
    int Lk, float scale) {
  __shared__ __align__(16) h16 Qs [64 * 72];  // [qrow][dh]
  __shared__ __align__(16) h16 Ks [64 * 72];  // [key ][dh]   (Bt layout for S)
  __shared__ __align__(16) h16 Vts[64 * 72];  // [dh  ][key]  (Bt layout for O)
  __shared__ __align__(16) h16 Ps [64 * 72];  // [qrow][key]  (A layout for O)

  const int tid = threadIdx.x, lane = tid & 31, wave = tid >> 5;
  const int h  = blockIdx.y;
  const int q0 = blockIdx.x * 64;
  const int co = h * DH;
  const int wr0 = wave * 16;
  const int hs = (lane < 16) ? 0 : 8;
  const int kb = (lane < 16) ? 0 : 8;

  // load Q tile 64x64
  {
    int r = tid >> 1, c0 = (tid & 1) * 32;
#pragma unroll
    for (int i = 0; i < 4; ++i)
      *(v8h*)(&Qs[r * 72 + c0 + i * 8]) =
          *(const v8h*)(Q + (size_t)(q0 + r) * CDIM + co + c0 + i * 8);
  }
  __syncthreads();

  v16h aq[2];
#pragma unroll
  for (int ks = 0; ks < 2; ++ks) {
    int r = wr0 + (lane & 15);
    aq[ks] = cat16(*(const v8h*)(&Qs[r * 72 + ks * 32 + hs]),
                   *(const v8h*)(&Qs[r * 72 + ks * 32 + hs + 16]));
  }

  v8f mst, lst, oacc[4];
#pragma unroll
  for (int g = 0; g < 8; ++g) { mst[g] = -3.0e38f; lst[g] = 0.0f; }
#pragma unroll
  for (int ni = 0; ni < 4; ++ni) oacc[ni] = zero8f();

  const int nblk = (Lk + 63) >> 6;
  for (int jb = 0; jb < nblk; ++jb) {
    const int j0 = jb * 64;
    __syncthreads();
    // K tile [key][dh] + V tile transposed [dh][key]
    {
      int r = tid >> 1, c0 = (tid & 1) * 32;
      bool ok = (j0 + r) < Lk;
#pragma unroll
      for (int i = 0; i < 4; ++i) {
        v8h t = zero8h();
        if (ok) t = *(const v8h*)(Kb + (size_t)(j0 + r) * CDIM + co + c0 + i * 8);
        *(v8h*)(&Ks[r * 72 + c0 + i * 8]) = t;
      }
#pragma unroll
      for (int i = 0; i < 4; ++i) {
        v8h t = zero8h();
        if (ok) t = *(const v8h*)(Vb + (size_t)(j0 + r) * CDIM + co + c0 + i * 8);
#pragma unroll
        for (int e = 0; e < 8; ++e) Vts[(c0 + i * 8 + e) * 72 + r] = t[e];
      }
    }
    __syncthreads();

    // S = Q * K^T  (16 q-rows x 64 keys per wave)
    v8f sacc[4];
#pragma unroll
    for (int ni = 0; ni < 4; ++ni) sacc[ni] = zero8f();
#pragma unroll
    for (int ni = 0; ni < 4; ++ni) {
#pragma unroll
      for (int ks = 0; ks < 2; ++ks) {
        int r = ni * 16 + (lane & 15);
        v16h bk = cat16(*(const v8h*)(&Ks[r * 72 + ks * 32 + kb]),
                        *(const v8h*)(&Ks[r * 72 + ks * 32 + kb + 16]));
        sacc[ni] = wmma_f16(aq[ks], bk, sacc[ni]);
      }
    }
    // scale + key mask + row max
    v8f cmax;
#pragma unroll
    for (int g = 0; g < 8; ++g) cmax[g] = -3.0e38f;
#pragma unroll
    for (int ni = 0; ni < 4; ++ni) {
      int j = j0 + ni * 16 + (lane & 15);
      bool valid = j < Lk;
#pragma unroll
      for (int g = 0; g < 8; ++g) {
        float s = sacc[ni][g] * scale;
        if (!valid) s = -3.0e38f;
        sacc[ni][g] = s;
        cmax[g] = fmaxf(cmax[g], s);
      }
    }
#pragma unroll
    for (int msk = 8; msk >= 1; msk >>= 1)
#pragma unroll
      for (int g = 0; g < 8; ++g)
        cmax[g] = fmaxf(cmax[g], __shfl_xor(cmax[g], msk, 32));

    v8f mnew, psc, rsum;
#pragma unroll
    for (int g = 0; g < 8; ++g) {
      mnew[g] = fmaxf(mst[g], cmax[g]);
      psc[g]  = expf(mst[g] - mnew[g]);
      rsum[g] = 0.0f;
    }
    // P = exp(S - mnew), bounce through LDS to re-shape C-layout -> A-layout
#pragma unroll
    for (int ni = 0; ni < 4; ++ni) {
#pragma unroll
      for (int g = 0; g < 8; ++g) {
        float p = expf(sacc[ni][g] - mnew[g]);
        rsum[g] += p;
        int rr = wr0 + ((lane < 16) ? g : g + 8);
        Ps[rr * 72 + ni * 16 + (lane & 15)] = (h16)p;
      }
    }
#pragma unroll
    for (int msk = 8; msk >= 1; msk >>= 1)
#pragma unroll
      for (int g = 0; g < 8; ++g) rsum[g] += __shfl_xor(rsum[g], msk, 32);
#pragma unroll
    for (int g = 0; g < 8; ++g) {
      lst[g] = lst[g] * psc[g] + rsum[g];
      mst[g] = mnew[g];
    }
#pragma unroll
    for (int ni = 0; ni < 4; ++ni)
#pragma unroll
      for (int g = 0; g < 8; ++g) oacc[ni][g] *= psc[g];

    // O += P * V  (wave-local P rows; same-wave LDS RAW handled by compiler waits)
#pragma unroll
    for (int ks = 0; ks < 2; ++ks) {
      int r = wr0 + (lane & 15);
      v16h ap = cat16(*(const v8h*)(&Ps[r * 72 + ks * 32 + hs]),
                      *(const v8h*)(&Ps[r * 72 + ks * 32 + hs + 16]));
#pragma unroll
      for (int ni = 0; ni < 4; ++ni) {
        int rv = ni * 16 + (lane & 15);
        v16h bv = cat16(*(const v8h*)(&Vts[rv * 72 + ks * 32 + kb]),
                        *(const v8h*)(&Vts[rv * 72 + ks * 32 + kb + 16]));
        oacc[ni] = wmma_f16(ap, bv, oacc[ni]);
      }
    }
  }

  // normalize + write O (f16 [N,512])
#pragma unroll
  for (int ni = 0; ni < 4; ++ni) {
    int col = co + ni * 16 + (lane & 15);
#pragma unroll
    for (int g = 0; g < 8; ++g) {
      int row = q0 + wr0 + ((lane < 16) ? g : g + 8);
      float v = oacc[ni][g] / lst[g];
      O[(size_t)row * CDIM + col] = (h16)v;
    }
  }
}

// ---------------------------------------------------------------------------
// Orchestration
// ---------------------------------------------------------------------------
extern "C" void kernel_launch(void* const* d_in, const int* in_sizes, int n_in,
                              void* d_out, int out_size, void* d_ws, size_t ws_size,
                              hipStream_t stream) {
  (void)in_sizes; (void)n_in; (void)out_size; (void)ws_size;
  const float* x       = (const float*)d_in[0];
  const float* context = (const float*)d_in[1];
  const float* gn_g  = (const float*)d_in[2];
  const float* gn_b  = (const float*)d_in[3];
  const float* pi_w  = (const float*)d_in[4];
  const float* pi_b  = (const float*)d_in[5];
  const float* po_w  = (const float*)d_in[6];
  const float* po_b  = (const float*)d_in[7];
  const float* q1    = (const float*)d_in[8];
  const float* k1    = (const float*)d_in[9];
  const float* v1    = (const float*)d_in[10];
  const float* o1_w  = (const float*)d_in[11];
  const float* o1_b  = (const float*)d_in[12];
  const float* q2    = (const float*)d_in[13];
  const float* k2    = (const float*)d_in[14];
  const float* v2    = (const float*)d_in[15];
  const float* o2_w  = (const float*)d_in[16];
  const float* o2_b  = (const float*)d_in[17];
  const float* ff_w1 = (const float*)d_in[18];
  const float* ff_b1 = (const float*)d_in[19];
  const float* ff_w2 = (const float*)d_in[20];
  const float* ff_b2 = (const float*)d_in[21];
  const float* ln1_g = (const float*)d_in[22];
  const float* ln1_b = (const float*)d_in[23];
  const float* ln2_g = (const float*)d_in[24];
  const float* ln2_b = (const float*)d_in[25];
  const float* ln3_g = (const float*)d_in[26];
  const float* ln3_b = (const float*)d_in[27];
  float* out = (float*)d_out;

  // ---- workspace carve-out (256B aligned) ----
  char* w = (char*)d_ws;
  size_t off = 0;
  auto alloc = [&](size_t bytes) -> void* {
    off = (off + 255) & ~(size_t)255;
    void* p = w + off;
    off += bytes;
    return p;
  };
  float* stats = (float*)alloc(64 * sizeof(float));
  h16*  T   = (h16*)alloc((size_t)NTOK * CDIM * 2);
  float* Y1 = (float*)alloc((size_t)NTOK * CDIM * 4);
  float* Y2 = (float*)alloc((size_t)NTOK * CDIM * 4);
  h16*  Qb  = (h16*)alloc((size_t)NTOK * CDIM * 2);
  h16*  Kbuf= (h16*)alloc((size_t)NTOK * CDIM * 2);
  h16*  Vbuf= (h16*)alloc((size_t)NTOK * CDIM * 2);
  h16*  Ob  = (h16*)alloc((size_t)NTOK * CDIM * 2);
  h16*  Hff = (h16*)alloc((size_t)NTOK * FFDIM * 2);
  h16*  ctx16 = (h16*)alloc((size_t)NCTX * CTXD * 2);
  h16* pi_wt = (h16*)alloc((size_t)CDIM * CDIM * 2);
  h16* po_wt = (h16*)alloc((size_t)CDIM * CDIM * 2);
  h16* q1t   = (h16*)alloc((size_t)CDIM * CDIM * 2);
  h16* k1t   = (h16*)alloc((size_t)CDIM * CDIM * 2);
  h16* v1t   = (h16*)alloc((size_t)CDIM * CDIM * 2);
  h16* o1t   = (h16*)alloc((size_t)CDIM * CDIM * 2);
  h16* q2t   = (h16*)alloc((size_t)CDIM * CDIM * 2);
  h16* o2t   = (h16*)alloc((size_t)CDIM * CDIM * 2);
  h16* k2t   = (h16*)alloc((size_t)CDIM * CTXD * 2);
  h16* v2t   = (h16*)alloc((size_t)CDIM * CTXD * 2);
  h16* ff1t  = (h16*)alloc((size_t)FFDIM * CDIM * 2);
  h16* ff2t  = (h16*)alloc((size_t)CDIM * FFDIM * 2);

  const int EW = (NTOK * CDIM + 255) / 256;         // elementwise grid over [N,C]
  const int WC5 = (CDIM * CDIM + 255) / 256;        // 512x512 weight cvt
  const int WCC = (CTXD * CDIM + 255) / 256;        // 768x512 weight cvt
  const int WCF = (CDIM * FFDIM + 255) / 256;       // 512x2048 weight cvt

  // ---- weight prep (fp32 [K,D] -> f16 [D,K]) ----
  wcvt_t_kernel<<<WC5, 256, 0, stream>>>(pi_w,  pi_wt, CDIM, CDIM);
  wcvt_t_kernel<<<WC5, 256, 0, stream>>>(po_w,  po_wt, CDIM, CDIM);
  wcvt_t_kernel<<<WC5, 256, 0, stream>>>(q1,    q1t,   CDIM, CDIM);
  wcvt_t_kernel<<<WC5, 256, 0, stream>>>(k1,    k1t,   CDIM, CDIM);
  wcvt_t_kernel<<<WC5, 256, 0, stream>>>(v1,    v1t,   CDIM, CDIM);
  wcvt_t_kernel<<<WC5, 256, 0, stream>>>(o1_w,  o1t,   CDIM, CDIM);
  wcvt_t_kernel<<<WC5, 256, 0, stream>>>(q2,    q2t,   CDIM, CDIM);
  wcvt_t_kernel<<<WC5, 256, 0, stream>>>(o2_w,  o2t,   CDIM, CDIM);
  wcvt_t_kernel<<<WCC, 256, 0, stream>>>(k2,    k2t,   CTXD, CDIM);
  wcvt_t_kernel<<<WCC, 256, 0, stream>>>(v2,    v2t,   CTXD, CDIM);
  wcvt_t_kernel<<<WCF, 256, 0, stream>>>(ff_w1, ff1t,  CDIM, FFDIM);
  wcvt_t_kernel<<<WCF, 256, 0, stream>>>(ff_w2, ff2t,  FFDIM, CDIM);
  cvt_f32_f16_kernel<<<(NCTX * CTXD + 255) / 256, 256, 0, stream>>>(
      context, ctx16, NCTX * CTXD);

  // ---- GroupNorm -> T (f16 [N,C]) ----
  gn_stats_kernel<<<32, 256, 0, stream>>>(x, stats);
  gn_apply_kernel<<<EW, 256, 0, stream>>>(x, gn_g, gn_b, stats, T);

  const dim3 gNC((NTOK + 127) / 128, CDIM / 64);    // M=4096, N=512
  const dim3 gFF((NTOK + 127) / 128, FFDIM / 64);   // M=4096, N=2048
  const dim3 gCTX(1, CDIM / 64);                    // M=77,   N=512
  const dim3 gAttn(NTOK / 64, HEADS);

  // proj_in:  Y1 = T @ pi_w + pi_b
  gemm_wmma_kernel<0, 1><<<gNC, 256, 0, stream>>>(
      T, pi_wt, pi_b, nullptr, nullptr, Y1, NTOK, CDIM, CDIM);

  // ---- self attention ----
  ln_kernel<<<NTOK, 256, 0, stream>>>(Y1, ln1_g, ln1_b, T);
  gemm_wmma_kernel<0, 0><<<gNC, 256, 0, stream>>>(
      T, q1t, nullptr, nullptr, Qb, nullptr, NTOK, CDIM, CDIM);
  gemm_wmma_kernel<0, 0><<<gNC, 256, 0, stream>>>(
      T, k1t, nullptr, nullptr, Kbuf, nullptr, NTOK, CDIM, CDIM);
  gemm_wmma_kernel<0, 0><<<gNC, 256, 0, stream>>>(
      T, v1t, nullptr, nullptr, Vbuf, nullptr, NTOK, CDIM, CDIM);
  attn_flash_kernel<<<gAttn, 128, 0, stream>>>(Qb, Kbuf, Vbuf, Ob, NTOK, 0.125f);
  gemm_wmma_kernel<0, 1><<<gNC, 256, 0, stream>>>(
      Ob, o1t, o1_b, Y1, nullptr, Y2, NTOK, CDIM, CDIM);

  // ---- cross attention ----
  ln_kernel<<<NTOK, 256, 0, stream>>>(Y2, ln2_g, ln2_b, T);
  gemm_wmma_kernel<0, 0><<<gNC, 256, 0, stream>>>(
      T, q2t, nullptr, nullptr, Qb, nullptr, NTOK, CDIM, CDIM);
  gemm_wmma_kernel<0, 0><<<gCTX, 256, 0, stream>>>(
      ctx16, k2t, nullptr, nullptr, Kbuf, nullptr, NCTX, CDIM, CTXD);
  gemm_wmma_kernel<0, 0><<<gCTX, 256, 0, stream>>>(
      ctx16, v2t, nullptr, nullptr, Vbuf, nullptr, NCTX, CDIM, CTXD);
  attn_flash_kernel<<<gAttn, 128, 0, stream>>>(Qb, Kbuf, Vbuf, Ob, NCTX, 0.125f);
  gemm_wmma_kernel<0, 1><<<gNC, 256, 0, stream>>>(
      Ob, o2t, o2_b, Y2, nullptr, Y1, NTOK, CDIM, CDIM);

  // ---- feed-forward (GELU) ----
  ln_kernel<<<NTOK, 256, 0, stream>>>(Y1, ln3_g, ln3_b, T);
  gemm_wmma_kernel<1, 0><<<gFF, 256, 0, stream>>>(
      T, ff1t, ff_b1, nullptr, Hff, nullptr, NTOK, FFDIM, CDIM);
  gemm_wmma_kernel<0, 1><<<gNC, 256, 0, stream>>>(
      Hff, ff2t, ff_b2, Y1, nullptr, Y2, NTOK, CDIM, FFDIM);

  // ---- proj_out (transposed store [C,N]) + input residual ----
  cvt_f32_f16_kernel<<<EW, 256, 0, stream>>>(Y2, T, NTOK * CDIM);
  gemm_wmma_kernel<0, 2><<<gNC, 256, 0, stream>>>(
      T, po_wt, po_b, x, nullptr, out, NTOK, CDIM, CDIM);
}
